// GemmaAttention_39616778338749
// MI455X (gfx1250) — matrix operand order repaired and checked by reading, not compiled, not run on hardware
//
#include <hip/hip_runtime.h>
#include <hip/hip_bf16.h>
#include <math.h>

// ---------------------------------------------------------------------------
// Gemma attention forward for gfx1250 (MI455X, wave32, WMMA).
// Matmuls: v_wmma_f32_16x16x32_bf16 (bf16 in, f32 accumulate).
// GEMM tiles staged with gfx1250 async global->LDS copies (ASYNCcnt).
// ---------------------------------------------------------------------------

#define S_LEN     2048
#define HID       3584
#define NUM_HEADS 16
#define NUM_KVH   8
#define HEAD_DIM  256
#define QKV_N     8192      // (16 + 2*8) * 256
#define Q_SIZE    4096
#define KV_SIZE   2048
#define ATTN_N    4096      // NUM_HEADS * HEAD_DIM
#define HALF_DIM  128
#define SCALING   0.0625f   // 256^-0.5
#define INV_SOFTCAP 0.02f   // 1/50
#define SOFTCAP   50.0f

typedef __attribute__((ext_vector_type(16))) __bf16 v16bf;
typedef __attribute__((ext_vector_type(8)))  float  v8f;

union FragU { uint4 u[2]; v16bf v; };

// f32 -> bf16 round-to-nearest-even, raw u16 storage
static __device__ __forceinline__ unsigned short f2bf(float f) {
    union { float f; unsigned int u; } x; x.f = f;
    unsigned int r = x.u + 0x7FFFu + ((x.u >> 16) & 1u);
    return (unsigned short)(r >> 16);
}
static __device__ __forceinline__ float bf2f(unsigned short b) {
    union { unsigned int u; float f; } x; x.u = ((unsigned int)b) << 16;
    return x.f;
}

static __device__ __forceinline__ v8f wmma_bf16(v16bf a, v16bf b, v8f c) {
    // v_wmma_f32_16x16x32_bf16  (neg_a, A, neg_b, B, c_mod, C, reuse_a, reuse_b)
    return __builtin_amdgcn_wmma_f32_16x16x32_bf16(false, a, false, b, (short)0, c, false, false);
}

// Async global->LDS 128-bit copy (per-lane addresses), tracked by ASYNCcnt.
static __device__ __forceinline__ void async_copy_b128(unsigned int lds_off,
                                                       const void* gaddr) {
    asm volatile("global_load_async_to_lds_b128 %0, %1, off"
                 :: "v"(lds_off), "v"((unsigned long long)(size_t)gaddr)
                 : "memory");
}
// Flat shared-pointer low 32 bits carry the LDS byte offset (aperture is in
// the high 32 bits per the CDNA5 flat-address aperture rules).
static __device__ __forceinline__ unsigned int lds_offset(const void* p) {
    return (unsigned int)(size_t)p;
}

// A-matrix fragment (16x32 bf16, M x K). p points at (row 0, k 0) of the tile,
// ld = row stride in elements. ISA layout: lane m=l&15, hi=l>>4;
// v[0..3] = k[hi*8 .. hi*8+7], v[4..7] = k[16+hi*8 .. 16+hi*8+7].
static __device__ __forceinline__ v16bf load_frag_A(const unsigned short* p, int ld, int lane) {
    int m = lane & 15, hi = lane >> 4;
    const uint4* q = (const uint4*)(p + (size_t)m * ld);
    FragU f; f.u[0] = q[hi]; f.u[1] = q[2 + hi];
    return f.v;
}

// B-matrix fragment (32x16 bf16, K x N) where B[k][n] = W[n][k] and W is
// row-major N x K. p points at (n 0, k 0), ld = W row stride.
// ISA layout: lane n=l&15, hi=l>>4; v[0..3] = k[hi*8 .. hi*8+7], v[4..7] = k[16+hi*8 .. 16+hi*8+7].
static __device__ __forceinline__ v16bf load_frag_B(const unsigned short* p, int ld, int lane) {
    int n = lane & 15, hi = lane >> 4;
    const uint4* q = (const uint4*)(p + (size_t)n * ld);
    FragU f; f.u[0] = q[hi]; f.u[1] = q[2 + hi];
    return f.v;
}

// ---------------------------------------------------------------------------
// 1) f32 -> bf16 cast
// ---------------------------------------------------------------------------
__global__ void cast_f32_to_bf16(const float* __restrict__ in,
                                 unsigned short* __restrict__ out, long n) {
    long i = (long)blockIdx.x * blockDim.x + threadIdx.x;
    if (i < n) out[i] = f2bf(in[i]);
}

// ---------------------------------------------------------------------------
// 2) GEMM  C[M,N] = A[M,K] * W[N,K]^T   (bf16 inputs, f32 accum)
//    block = 256 threads = 8 waves; tile 128(M) x 128(N); K-step 32.
//    Double-buffered async global->LDS staging; wave computes 32x64
//    (2 A-frags x 4 B-frags = 8 WMMAs per K-step).
// ---------------------------------------------------------------------------
__global__ __launch_bounds__(256) void gemm_bf16_nt(
        const unsigned short* __restrict__ A,
        const unsigned short* __restrict__ W,
        void* __restrict__ C,
        int M, int N, int K, int outBf16) {
    __shared__ __align__(16) unsigned short sA[2][128 * 32];
    __shared__ __align__(16) unsigned short sW[2][128 * 32];

    const int tid   = threadIdx.x;
    const int lane  = tid & 31;
    const int wid   = tid >> 5;
    const int wm    = wid & 3;   // 4 waves in M (32 rows each)
    const int wn    = wid >> 2;  // 2 waves in N (64 cols each)
    const int mBase = blockIdx.y * 128;
    const int nBase = blockIdx.x * 128;

    v8f acc[2][4];
#pragma unroll
    for (int mt = 0; mt < 2; ++mt)
#pragma unroll
        for (int nt = 0; nt < 4; ++nt)
#pragma unroll
            for (int j = 0; j < 8; ++j) acc[mt][nt][j] = 0.0f;

    // Issue one K-tile (128x32 of A and of W) as async b128 copies.
    // 512 uint4 per matrix -> each thread issues 2 + 2.
    auto issue_tile = [&](int buf, int kb) {
#pragma unroll
        for (int rep = 0; rep < 2; ++rep) {
            const int u   = tid + rep * 256;     // uint4 index within tile
            const int row = u >> 2;
            const int seg = u & 3;
            async_copy_b128(lds_offset(&sA[buf][0]) + u * 16,
                            A + (size_t)(mBase + row) * K + kb + seg * 8);
            async_copy_b128(lds_offset(&sW[buf][0]) + u * 16,
                            W + (size_t)(nBase + row) * K + kb + seg * 8);
        }
    };

    const int nk = K >> 5;
    issue_tile(0, 0);

    for (int it = 0; it < nk; ++it) {
        const int cur = it & 1;
        const bool more = (it + 1) < nk;
        if (more) issue_tile(cur ^ 1, (it + 1) << 5);
        // Async loads complete in order: waiting <=4 releases the older group.
        if (more) asm volatile("s_wait_asynccnt 0x4" ::: "memory");
        else      asm volatile("s_wait_asynccnt 0x0" ::: "memory");
        __syncthreads();   // current buffer fully resident for all waves

        v16bf a0 = load_frag_A(&sA[cur][(wm * 32) * 32], 32, lane);
        v16bf a1 = load_frag_A(&sA[cur][(wm * 32 + 16) * 32], 32, lane);
#pragma unroll
        for (int nt = 0; nt < 4; ++nt) {
            v16bf b = load_frag_B(&sW[cur][(wn * 64 + nt * 16) * 32], 32, lane);
            acc[0][nt] = wmma_bf16(a0, b, acc[0][nt]);
            acc[1][nt] = wmma_bf16(a1, b, acc[1][nt]);
        }
        __syncthreads();   // everyone done reading cur before it is re-filled
    }

#pragma unroll
    for (int mt = 0; mt < 2; ++mt) {
        const int m0 = mBase + wm * 32 + mt * 16 + (lane >> 4) * 8;
#pragma unroll
        for (int nt = 0; nt < 4; ++nt) {
            const int n = nBase + wn * 64 + nt * 16 + (lane & 15);
#pragma unroll
            for (int j = 0; j < 8; ++j) {
                size_t idx = (size_t)(m0 + j) * N + n;
                if (outBf16) ((unsigned short*)C)[idx] = f2bf(acc[mt][nt][j]);
                else         ((float*)C)[idx]          = acc[mt][nt][j];
            }
        }
    }
}

// ---------------------------------------------------------------------------
// 3) RoPE + split into Q (scaled), K, V^T buffers (all bf16)
//    qkv layout per row s: [Q 16*256 | K 8*256 | V 8*256]
// ---------------------------------------------------------------------------
__global__ void rope_split_kernel(const unsigned short* __restrict__ qkv,
                                  const float* __restrict__ cosT,
                                  const float* __restrict__ sinT,
                                  unsigned short* __restrict__ qbuf,  // [16][2048][256]
                                  unsigned short* __restrict__ kbuf,  // [8][2048][256]
                                  unsigned short* __restrict__ vT) {  // [8][256][2048]
    long i = (long)blockIdx.x * blockDim.x + threadIdx.x;
    if (i >= (long)S_LEN * QKV_N) return;
    const int s   = (int)(i >> 13);
    const int col = (int)(i & (QKV_N - 1));

    if (col < Q_SIZE) {
        const int h = col >> 8, d = col & 255, dd = d & 127;
        const unsigned short* base = qkv + ((size_t)s << 13) + h * 256;
        float xr = bf2f(base[dd]);
        float xi = bf2f(base[dd + HALF_DIM]);
        float c  = cosT[s * HALF_DIM + dd];
        float sn = sinT[s * HALF_DIM + dd];
        float v  = (d < HALF_DIM) ? (xr * c - xi * sn) : (xr * sn + xi * c);
        qbuf[((size_t)h * S_LEN + s) * HEAD_DIM + d] = f2bf(v * SCALING);
    } else if (col < Q_SIZE + KV_SIZE) {
        const int kc = col - Q_SIZE;
        const int kh = kc >> 8, d = kc & 255, dd = d & 127;
        const unsigned short* base = qkv + ((size_t)s << 13) + Q_SIZE + kh * 256;
        float xr = bf2f(base[dd]);
        float xi = bf2f(base[dd + HALF_DIM]);
        float c  = cosT[s * HALF_DIM + dd];
        float sn = sinT[s * HALF_DIM + dd];
        float v  = (d < HALF_DIM) ? (xr * c - xi * sn) : (xr * sn + xi * c);
        kbuf[((size_t)kh * S_LEN + s) * HEAD_DIM + d] = f2bf(v);
    } else {
        const int vc = col - (Q_SIZE + KV_SIZE);
        const int kh = vc >> 8, d = vc & 255;
        vT[(size_t)kh * HEAD_DIM * S_LEN + (size_t)d * S_LEN + s] = qkv[i];
    }
}

// ---------------------------------------------------------------------------
// 4) Flash attention, one wave per (head, 16-row q block).
//    scores = Q Kt (WMMA), tanh softcap, causal, online softmax, P V (WMMA).
// ---------------------------------------------------------------------------
__global__ __launch_bounds__(32) void attn_kernel(
        const unsigned short* __restrict__ qbuf,
        const unsigned short* __restrict__ kbuf,
        const unsigned short* __restrict__ vT,
        unsigned short* __restrict__ attn_out) {   // [2048][4096] bf16
    __shared__ __align__(16) unsigned short sQ[16 * 256];
    __shared__ __align__(16) unsigned short sP[16 * 32];

    const int qb   = blockIdx.x;       // 0..127
    const int h    = blockIdx.y;       // 0..15
    const int kh   = h >> 1;           // GQA: 2 q-heads share a kv-head
    const int lane = threadIdx.x;

    // stage 16x256 bf16 Q tile into LDS
    const unsigned short* qg = qbuf + ((size_t)h * S_LEN + qb * 16) * HEAD_DIM;
#pragma unroll
    for (int r = 0; r < 16; ++r)
        ((uint4*)sQ)[r * 32 + lane] = ((const uint4*)(qg + r * HEAD_DIM))[lane];

    v8f acc[16];
#pragma unroll
    for (int dt = 0; dt < 16; ++dt)
#pragma unroll
        for (int j = 0; j < 8; ++j) acc[dt][j] = 0.0f;

    float mstate[8], lstate[8];
#pragma unroll
    for (int j = 0; j < 8; ++j) { mstate[j] = -1e30f; lstate[j] = 0.0f; }

    const int mrow0 = qb * 16 + (lane >> 4) * 8;       // + j gives global row
    const int nkb   = ((qb * 16 + 15) >> 5) + 1;       // causal: 32-key blocks
    const unsigned short* kg = kbuf + (size_t)kh * S_LEN * HEAD_DIM;
    const unsigned short* vg = vT   + (size_t)kh * HEAD_DIM * S_LEN;

    for (int kb2 = 0; kb2 < nkb; ++kb2) {
        const int kbase = kb2 * 32;

        // L2 prefetch of the next key block (K rows + V^T columns)
        if (kb2 + 1 < nkb) {
            __builtin_prefetch(kg + (size_t)(kbase + 32 + lane) * HEAD_DIM, 0, 1);
            __builtin_prefetch(vg + (size_t)(lane * 8) * S_LEN + kbase + 32, 0, 1);
        }

        // scores for two 16-key halves, 8 chained WMMAs each over d=256
        v8f sc[2];
#pragma unroll
        for (int half = 0; half < 2; ++half) {
            v8f c;
#pragma unroll
            for (int j = 0; j < 8; ++j) c[j] = 0.0f;
            const unsigned short* krow = kg + (size_t)(kbase + half * 16) * HEAD_DIM;
#pragma unroll
            for (int dc = 0; dc < 8; ++dc) {
                v16bf a = load_frag_A(sQ + dc * 32, 256, lane);
                v16bf b = load_frag_B(krow + dc * 32, 256, lane);
                c = wmma_bf16(a, b, c);
            }
            sc[half] = c;
        }

        __syncthreads();   // sP reuse from previous iteration

        // softcap + causal mask + online softmax statistics
#pragma unroll
        for (int j = 0; j < 8; ++j) {
            const int mg = mrow0 + j;
            float s0 = SOFTCAP * tanhf(sc[0][j] * INV_SOFTCAP);
            float s1 = SOFTCAP * tanhf(sc[1][j] * INV_SOFTCAP);
            const int n0 = kbase + (lane & 15);
            if (n0 > mg)      s0 += -1.0e9f;
            if (n0 + 16 > mg) s1 += -1.0e9f;

            float rm = fmaxf(s0, s1);
#pragma unroll
            for (int off = 1; off < 16; off <<= 1)
                rm = fmaxf(rm, __shfl_xor(rm, off, 16));
            const float mnew  = fmaxf(mstate[j], rm);
            const float alpha = expf(mstate[j] - mnew);
            const float p0 = expf(s0 - mnew);
            const float p1 = expf(s1 - mnew);
            float rs = p0 + p1;
#pragma unroll
            for (int off = 1; off < 16; off <<= 1)
                rs += __shfl_xor(rs, off, 16);
            lstate[j] = lstate[j] * alpha + rs;
            mstate[j] = mnew;
#pragma unroll
            for (int dt = 0; dt < 16; ++dt) acc[dt][j] *= alpha;

            const int pr = (lane >> 4) * 8 + j;
            sP[pr * 32 + (lane & 15)]      = f2bf(p0);
            sP[pr * 32 + 16 + (lane & 15)] = f2bf(p1);
        }

        __syncthreads();   // P visible before re-fragmenting

        // acc += P (16x32) * V (32x256) ; vT gives contiguous key pairs
#pragma unroll
        for (int dt = 0; dt < 16; ++dt) {
            v16bf a = load_frag_A(sP, 32, lane);
            v16bf b = load_frag_B(vg + (size_t)(dt * 16) * S_LEN + kbase, S_LEN, lane);
            acc[dt] = wmma_bf16(a, b, acc[dt]);
        }
    }

    // normalize and store bf16 [row][h*256 + d]
#pragma unroll
    for (int j = 0; j < 8; ++j) {
        const float inv = 1.0f / lstate[j];
        const size_t rowb = (size_t)(mrow0 + j) * ATTN_N;
#pragma unroll
        for (int dt = 0; dt < 16; ++dt) {
            const int colg = h * HEAD_DIM + dt * 16 + (lane & 15);
            attn_out[rowb + colg] = f2bf(acc[dt][j] * inv);
        }
    }
}

// ---------------------------------------------------------------------------
// host-side orchestration
// ---------------------------------------------------------------------------
static inline size_t align256(size_t x) { return (x + 255) & ~(size_t)255; }

extern "C" void kernel_launch(void* const* d_in, const int* in_sizes, int n_in,
                              void* d_out, int out_size, void* d_ws, size_t ws_size,
                              hipStream_t stream) {
    (void)in_sizes; (void)n_in; (void)out_size; (void)ws_size;

    const float* hidden = (const float*)d_in[0];   // [1,2048,3584]
    const float* fcos   = (const float*)d_in[1];   // [2048,128]
    const float* fsin   = (const float*)d_in[2];   // [2048,128]
    /* d_in[3] = mask: causal mask computed arithmetically in-kernel */
    const float* qkv_w  = (const float*)d_in[4];   // [8192,3584]
    const float* o_w    = (const float*)d_in[5];   // [3584,4096]

    char* ws = (char*)d_ws;
    size_t off = 0;
    unsigned short* hid_bf  = (unsigned short*)(ws + off); off = align256(off + (size_t)S_LEN * HID * 2);
    unsigned short* qkvw_bf = (unsigned short*)(ws + off); off = align256(off + (size_t)QKV_N * HID * 2);
    unsigned short* ow_bf   = (unsigned short*)(ws + off); off = align256(off + (size_t)HID * ATTN_N * 2);
    unsigned short* qkv_bf  = (unsigned short*)(ws + off); off = align256(off + (size_t)S_LEN * QKV_N * 2);
    unsigned short* q_bf    = (unsigned short*)(ws + off); off = align256(off + (size_t)NUM_HEADS * S_LEN * HEAD_DIM * 2);
    unsigned short* k_bf    = (unsigned short*)(ws + off); off = align256(off + (size_t)NUM_KVH * S_LEN * HEAD_DIM * 2);
    unsigned short* vT_bf   = (unsigned short*)(ws + off); off = align256(off + (size_t)NUM_KVH * HEAD_DIM * S_LEN * 2);
    unsigned short* attn_bf = (unsigned short*)(ws + off); off = align256(off + (size_t)S_LEN * ATTN_N * 2);

    // 1) casts
    {
        long n0 = (long)S_LEN * HID;
        cast_f32_to_bf16<<<(int)((n0 + 255) / 256), 256, 0, stream>>>(hidden, hid_bf, n0);
        long n1 = (long)QKV_N * HID;
        cast_f32_to_bf16<<<(int)((n1 + 255) / 256), 256, 0, stream>>>(qkv_w, qkvw_bf, n1);
        long n2 = (long)HID * ATTN_N;
        cast_f32_to_bf16<<<(int)((n2 + 255) / 256), 256, 0, stream>>>(o_w, ow_bf, n2);
    }

    // 2) QKV GEMM: [2048,3584] x [8192,3584]^T -> bf16 [2048,8192]
    {
        dim3 grid(QKV_N / 128, S_LEN / 128);
        gemm_bf16_nt<<<grid, 256, 0, stream>>>(hid_bf, qkvw_bf, qkv_bf,
                                               S_LEN, QKV_N, HID, /*outBf16=*/1);
    }

    // 3) RoPE + split (V stored transposed for WMMA B fragments)
    {
        long n = (long)S_LEN * QKV_N;
        rope_split_kernel<<<(int)((n + 255) / 256), 256, 0, stream>>>(
            qkv_bf, fcos, fsin, q_bf, k_bf, vT_bf);
    }

    // 4) flash attention, wave per (16-row q block, head)
    {
        dim3 grid(S_LEN / 16, NUM_HEADS);
        attn_kernel<<<grid, 32, 0, stream>>>(q_bf, k_bf, vT_bf, attn_bf);
    }

    // 5) O projection: [2048,4096] x [3584,4096]^T -> f32 d_out [2048,3584]
    {
        dim3 grid(HID / 128, S_LEN / 128);
        gemm_bf16_nt<<<grid, 256, 0, stream>>>(attn_bf, ow_bf, d_out,
                                               S_LEN, HID, ATTN_N, /*outBf16=*/0);
    }
}
